// GDSKR_85950885527942
// MI455X (gfx1250) — matrix operand order repaired and checked by reading, not compiled, not run on hardware
//
#include <hip/hip_runtime.h>
#include <hip/hip_bf16.h>
#include <math.h>

typedef __attribute__((ext_vector_type(16))) _Float16 v16h;
typedef __attribute__((ext_vector_type(8)))  _Float16 v8h;
typedef __attribute__((ext_vector_type(8)))  float    v8f;

typedef unsigned int u32x4 __attribute__((ext_vector_type(4)));
typedef int          s32x4 __attribute__((ext_vector_type(4)));
typedef int          s32x8 __attribute__((ext_vector_type(8)));

#define B_  8
#define Q_  2048
#define K_  4096
#define DIN 15

#if defined(__gfx1250__) && __has_builtin(__builtin_amdgcn_tensor_load_to_lds) && \
    __has_builtin(__builtin_amdgcn_s_wait_tensorcnt)
#define HAVE_TDM 1
#else
#define HAVE_TDM 0
#endif

// tanh-approx GELU (jax.nn.gelu approximate=True). Native V_TANH_F32 on gfx1250.
__device__ __forceinline__ float gelu_tanh(float x) {
  const float c0 = 0.7978845608028654f;
  const float c1 = 0.044715f;
  float u = c0 * (x + c1 * x * x * x);
#if defined(__gfx1250__) && __has_builtin(__builtin_amdgcn_tanhf)
  float th = __builtin_amdgcn_tanhf(u);
#else
  float e = __expf(2.0f * u);          // branch-free fallback: v_exp_f32
  float th = 1.0f - 2.0f / (e + 1.0f);
#endif
  return 0.5f * x * (1.0f + th);
}

// Load a 16x32 f16 A-fragment slice for this lane from a row-major row pointer.
// Contiguous K order: lane holds row (lane%16), K = {off..off+15}.
__device__ __forceinline__ v16h ldA16(const _Float16* p, int off) {
  v8h lo = *(const v8h*)(p + off);
  v8h hi = *(const v8h*)(p + off + 8);
  return __builtin_shufflevector(lo, hi, 0,1,2,3,4,5,6,7,8,9,10,11,12,13,14,15);
}

#if HAVE_TDM
// Issue one TDM DMA: copy `elems` dwords from gaddr -> LDS[ldsAddr].
// D# built per CDNA5 ISA:
//   g0: count=1, lds_addr, global_addr[56:0], type=2
//   g1: data_size=4B, tensor_dim0=elems, tensor_dim1=1, tile_dim0=elems,
//       tile_dim1=1, stride0=stride1=elems
__device__ __forceinline__ void tdm_load_tile(unsigned ldsAddr,
                                              unsigned long long gaddr,
                                              unsigned elems) {
  u32x4 g0;
  g0[0] = 1u;                                             // count=1 (valid)
  g0[1] = ldsAddr;                                        // lds_addr (bytes)
  g0[2] = (unsigned)gaddr;                                // global_addr lo32
  g0[3] = (unsigned)((gaddr >> 32) & 0x1FFFFFFu) | (2u << 30);  // hi bits|type=2
  s32x8 g1;
  g1[0] = (int)(2u << 16);                                // data_size = 4 bytes
  g1[1] = (int)((elems & 0xFFFFu) << 16);                 // tensor_dim0 lo16
  g1[2] = (int)(((elems >> 16) & 0xFFFFu) | (1u << 16));  // dim0 hi16 | dim1=1
  g1[3] = (int)((elems & 0xFFFFu) << 16);                 // tile_dim0
  g1[4] = 1;                                              // tile_dim1=1
  g1[5] = (int)elems;                                     // dim0_stride lo32
  g1[6] = (int)((elems & 0xFFFFu) << 16);                 // dim1_stride lo16
  g1[7] = 0;
  s32x4 g2 = {0, 0, 0, 0};
  s32x4 g3 = {0, 0, 0, 0};
#if defined(__clang_major__) && __clang_major__ >= 23
  s32x8 g4 = {0, 0, 0, 0, 0, 0, 0, 0};
  __builtin_amdgcn_tensor_load_to_lds(g0, g1, g2, g3, g4, 0);
#else
  __builtin_amdgcn_tensor_load_to_lds(g0, g1, g2, g3, 0);
#endif
}
#endif

// ---------------------------------------------------------------------------
// Weight prep: transpose f32 [K][N] -> f16 [N][K] (W1 K-padded 15 -> 32)
// ---------------------------------------------------------------------------
__global__ void prep_weights(const float* __restrict__ W1,
                             const float* __restrict__ W2,
                             const float* __restrict__ W3,
                             _Float16* __restrict__ W1T,
                             _Float16* __restrict__ W2T,
                             _Float16* __restrict__ W3T) {
  int i = blockIdx.x * blockDim.x + threadIdx.x;
  if (i < 256 * 32) {
    int n = i >> 5, k = i & 31;
    W1T[i] = (k < DIN) ? (_Float16)W1[k * 256 + n] : (_Float16)0.0f;
  } else if (i < 256 * 32 + 128 * 256) {
    int j = i - 256 * 32;
    int n = j >> 8, k = j & 255;
    W2T[j] = (_Float16)W2[k * 128 + n];
  } else if (i < 256 * 32 + 128 * 256 + 64 * 128) {
    int j = i - (256 * 32 + 128 * 256);
    int n = j >> 7, k = j & 127;
    W3T[j] = (_Float16)W3[k * 64 + n];
  }
}

// ---------------------------------------------------------------------------
// Fused MLP(15->256 gelu ->128 gelu ->64) + LayerNorm, one wave per 16 rows.
// All matmuls via v_wmma_f32_16x16x32_f16, activations staged through LDS.
// ---------------------------------------------------------------------------
__global__ __launch_bounds__(32)
void mlp_ln_kernel(const float* __restrict__ s_src,      // [rows][3]
                   const float* __restrict__ f_src,      // [rows][8] or nullptr
                   const float* __restrict__ emb_row,    // 4 floats
                   const float* __restrict__ b1,
                   const float* __restrict__ b2,
                   const float* __restrict__ b3,
                   const float* __restrict__ ln_s,
                   const float* __restrict__ ln_b,
                   const _Float16* __restrict__ W1T,     // [256][32]
                   const _Float16* __restrict__ W2T,     // [128][256]
                   const _Float16* __restrict__ W3T,     // [64][128]
                   float* __restrict__ out)              // [rows][64]
{
  __shared__ __align__(16) _Float16 Xs[16 * 40];
  __shared__ __align__(16) _Float16 H1[16 * 264];
  __shared__ __align__(16) _Float16 H2[16 * 136];
  __shared__ __align__(16) float    O [16 * 68];

  const int lane    = threadIdx.x;
  const int rowbase = blockIdx.x * 16;

  // Build padded input tile [16][32] f16 in LDS (lanes 0..15, one row each).
  if (lane < 16) {
    int g = rowbase + lane;
    const float* sp = s_src + (size_t)g * 3;
    float f[DIN];
#pragma unroll
    for (int j = 0; j < 4; ++j) f[j] = emb_row[j];
#pragma unroll
    for (int j = 0; j < 3; ++j) f[4 + j] = sp[j];
    if (f_src) {
      const float* fp = f_src + (size_t)g * 8;
#pragma unroll
      for (int j = 0; j < 8; ++j) f[7 + j] = fp[j];
    } else {
#pragma unroll
      for (int j = 0; j < 8; ++j) f[7 + j] = 0.0f;
    }
#pragma unroll
    for (int j = 0; j < 32; ++j)
      Xs[lane * 40 + j] = (j < DIN) ? (_Float16)f[j] : (_Float16)0.0f;
  }
  __syncthreads();

  const int row = lane & 15;           // A row / B,C,D column within tile
  const int kb  = (lane < 16) ? 0 : 16; // A-fragment K sub-offset
  const int kw  = (lane < 16) ? 0 : 16;// B-fragment K sub-offset
  const int mb  = (lane < 16) ? 0 : 8; // C/D M base

  // ---- Layer 1: [16x32] x [32x256] -> gelu -> H1[16][256]
  {
    v16h a = ldA16(Xs + row * 40, kb);
    for (int t = 0; t < 16; ++t) {
      int n = t * 16 + row;
      v16h bm = *(const v16h*)(W1T + n * 32 + kw);
      v8f c = {};
      c = __builtin_amdgcn_wmma_f32_16x16x32_f16(false, a, false, bm,
                                                 (short)0, c, false, false);
      float bias = b1[n];
#pragma unroll
      for (int v = 0; v < 8; ++v)
        H1[(mb + v) * 264 + n] = (_Float16)gelu_tanh(c[v] + bias);
    }
  }
  __syncthreads();

  // ---- Layer 2: [16x256] x [256x128] -> gelu -> H2[16][128]
  {
    v16h a2[8];
#pragma unroll
    for (int kc = 0; kc < 8; ++kc)
      a2[kc] = ldA16(H1 + row * 264, kc * 32 + kb);
    for (int t = 0; t < 8; ++t) {
      int n = t * 16 + row;
      const _Float16* wp = W2T + (size_t)n * 256 + kw;
      __builtin_prefetch(wp + 256, 0, 3);   // prefetch of W2T row n+1
      v8f c = {};
#pragma unroll
      for (int kc = 0; kc < 8; ++kc) {
        v16h bm = *(const v16h*)(wp + kc * 32);
        c = __builtin_amdgcn_wmma_f32_16x16x32_f16(false, a2[kc], false, bm,
                                                   (short)0, c, false, false);
      }
      float bias = b2[n];
#pragma unroll
      for (int v = 0; v < 8; ++v)
        H2[(mb + v) * 136 + n] = (_Float16)gelu_tanh(c[v] + bias);
    }
  }
  __syncthreads();

  // ---- Layer 3: [16x128] x [128x64] -> O[16][64] (f32, +bias, no act)
  {
    v16h a3[4];
#pragma unroll
    for (int kc = 0; kc < 4; ++kc)
      a3[kc] = ldA16(H2 + row * 136, kc * 32 + kb);
    for (int t = 0; t < 4; ++t) {
      int n = t * 16 + row;
      const _Float16* wp = W3T + (size_t)n * 128 + kw;
      v8f c = {};
#pragma unroll
      for (int kc = 0; kc < 4; ++kc) {
        v16h bm = *(const v16h*)(wp + kc * 32);
        c = __builtin_amdgcn_wmma_f32_16x16x32_f16(false, a3[kc], false, bm,
                                                   (short)0, c, false, false);
      }
      float bias = b3[n];
#pragma unroll
      for (int v = 0; v < 8; ++v)
        O[(mb + v) * 68 + n] = c[v] + bias;
    }
  }
  __syncthreads();

  // ---- LayerNorm(64): 2 lanes per row, pair-reduce via wave32 shuffle.
  {
    int r = lane >> 1;
    int h = lane & 1;
    const float* op = O + r * 68 + h * 32;
    float s = 0.f, ss = 0.f;
#pragma unroll
    for (int j = 0; j < 32; ++j) { float x = op[j]; s += x; ss += x * x; }
    s  += __shfl_xor(s, 1, 32);
    ss += __shfl_xor(ss, 1, 32);
    float mean = s * (1.0f / 64.0f);
    float var  = ss * (1.0f / 64.0f) - mean * mean;
    float rinv = rsqrtf(var + 1e-6f);
    int g = rowbase + r;
    float* po = out + (size_t)g * 64 + h * 32;
#pragma unroll
    for (int j = 0; j < 32; ++j) {
      int col = h * 32 + j;
      po[j] = (op[j] - mean) * rinv * ln_s[col] + ln_b[col];
    }
  }
}

// ---------------------------------------------------------------------------
// Brute-force kNN (k=16, 3-D), candidate tiles DMA'd to LDS by the Tensor
// Data Mover (double-buffered: TDM for tile t+1 overlaps selection of tile t).
// One thread per query; 256 queries (same batch) per block.
// ---------------------------------------------------------------------------
__global__ __launch_bounds__(256)
void knn_kernel(const float* __restrict__ rx, int nr,
                const float* __restrict__ tx, int nt,
                float* __restrict__ dOut, float* __restrict__ iOut)
{
  __shared__ __align__(16) float tile[2][512 * 3];   // 2 x 6KB, point-major xyz
  int blocksPerBatch = nr >> 8;
  int b  = blockIdx.x / blocksPerBatch;
  int qi = (blockIdx.x % blocksPerBatch) * 256 + threadIdx.x;

  const float* rp = rx + ((size_t)b * nr + qi) * 3;
  float qx = rp[0], qy = rp[1], qz = rp[2];

  float bd[16]; int bi[16];
#pragma unroll
  for (int j = 0; j < 16; ++j) { bd[j] = 3.4e38f; bi[j] = 0; }

  const float* tb = tx + (size_t)b * nt * 3;
  const int ntiles = nt >> 9;                        // nt / 512

#if HAVE_TDM
  const unsigned long long gbase = (unsigned long long)(uintptr_t)tb;
  const unsigned ldsbase = (unsigned)(uintptr_t)&tile[0][0];
  if (threadIdx.x < 32)                              // wave 0 drives the TDM
    tdm_load_tile(ldsbase, gbase, 1536u);
#endif

  for (int t = 0; t < ntiles; ++t) {
    const float* buf = &tile[t & 1][0];
#if HAVE_TDM
    if (threadIdx.x < 32) {
      if (t + 1 < ntiles) {
        tdm_load_tile(ldsbase + (unsigned)((t + 1) & 1) * 6144u,
                      gbase + (unsigned long long)(t + 1) * 6144ull, 1536u);
        __builtin_amdgcn_s_wait_tensorcnt(1);        // tile t landed in LDS
      } else {
        __builtin_amdgcn_s_wait_tensorcnt(0);
      }
    }
    __syncthreads();                                 // tile t visible to block
#else
    for (int i = threadIdx.x; i < 512; i += 256) {
      const float* tp = tb + (size_t)(t * 512 + i) * 3;
      float* dst = &tile[t & 1][i * 3];
      dst[0] = tp[0]; dst[1] = tp[1]; dst[2] = tp[2];
    }
    __syncthreads();
#endif
    for (int i = 0; i < 512; ++i) {
      float dx = qx - buf[i * 3 + 0];
      float dy = qy - buf[i * 3 + 1];
      float dz = qz - buf[i * 3 + 2];
      float d2 = dx * dx + dy * dy + dz * dz;
      if (d2 < bd[15]) {               // guarded insertion into sorted list
        float cd = d2; int ci = t * 512 + i;
#pragma unroll
        for (int j = 0; j < 16; ++j) {
          bool  sm = cd < bd[j];
          float od = bd[j]; int oi = bi[j];
          bd[j] = sm ? cd : od;  bi[j] = sm ? ci : oi;
          cd    = sm ? od : cd;  ci    = sm ? oi : ci;
        }
      }
    }
    __syncthreads();                   // all reads of buf done before refill
  }

  size_t ob = ((size_t)b * nr + qi) * 16;
#pragma unroll
  for (int j = 0; j < 16; ++j) {
    dOut[ob + j] = sqrtf(bd[j]);
    iOut[ob + j] = (float)bi[j];       // indices stored as f32 in flat output
  }
}

// ---------------------------------------------------------------------------
extern "C" void kernel_launch(void* const* d_in, const int* in_sizes, int n_in,
                              void* d_out, int out_size, void* d_ws, size_t ws_size,
                              hipStream_t stream) {
  (void)in_sizes; (void)n_in; (void)out_size; (void)ws_size;
  const float* s_ctx  = (const float*)d_in[0];
  const float* f_ctx  = (const float*)d_in[1];
  const float* s_test = (const float*)d_in[2];
  const float* emb    = (const float*)d_in[3];
  const float* W1 = (const float*)d_in[4];
  const float* b1 = (const float*)d_in[5];
  const float* W2 = (const float*)d_in[6];
  const float* b2 = (const float*)d_in[7];
  const float* W3 = (const float*)d_in[8];
  const float* b3 = (const float*)d_in[9];
  const float* ln_s = (const float*)d_in[10];
  const float* ln_b = (const float*)d_in[11];
  float* out = (float*)d_out;

  // Workspace: f16 transposed weights ([N][K] layout feeds B-fragments as
  // one contiguous 32B load per lane).
  _Float16* W1T = (_Float16*)d_ws;           // 256*32
  _Float16* W2T = W1T + 256 * 32;            // 128*256
  _Float16* W3T = W2T + 128 * 256;           // 64*128

  int prep_elems = 256 * 32 + 128 * 256 + 64 * 128; // 49152
  prep_weights<<<(prep_elems + 255) / 256, 256, 0, stream>>>(W1, W2, W3,
                                                             W1T, W2T, W3T);

  // Output layout (flat f32, reference return order):
  float* x_test = out;
  float* x_ctx  = x_test + (size_t)B_ * Q_ * 64;
  float* d_tc   = x_ctx  + (size_t)B_ * K_ * 64;
  float* i_tc   = d_tc   + (size_t)B_ * Q_ * 16;
  float* d_cc   = i_tc   + (size_t)B_ * Q_ * 16;
  float* i_cc   = d_cc   + (size_t)B_ * K_ * 16;

  // MLP+LN: emb row 0 = unobserved (test), row 1 = observed (ctx).
  mlp_ln_kernel<<<B_ * Q_ / 16, 32, 0, stream>>>(
      s_test, nullptr, emb + 0, b1, b2, b3, ln_s, ln_b, W1T, W2T, W3T, x_test);
  mlp_ln_kernel<<<B_ * K_ / 16, 32, 0, stream>>>(
      s_ctx, f_ctx, emb + 4, b1, b2, b3, ln_s, ln_b, W1T, W2T, W3T, x_ctx);

  // kNN graphs.
  knn_kernel<<<B_ * Q_ / 256, 256, 0, stream>>>(s_test, Q_, s_ctx, K_, d_tc, i_tc);
  knn_kernel<<<B_ * K_ / 256, 256, 0, stream>>>(s_ctx,  K_, s_ctx, K_, d_cc, i_cc);
}
